// TernBinLayer_37752762532729
// MI455X (gfx1250) — matrix operand order repaired and checked by reading, not compiled, not run on hardware
//
#include <hip/hip_runtime.h>

// Problem constants (match reference)
#define B_ROWS 8192
#define DIN    4096
#define DOUT   4096
#define KSTEP  64               // LDS stage depth (two WMMA K=32 chunks)
#define NK     (DIN / KSTEP)    // 64 K iterations

// 128x128 C tile per workgroup; 8 waves in a 4(M) x 2(N) grid,
// each wave computes 32(M) x 64(N) via 2x4 WMMA 16x16 tiles.
#define TILE_M 128
#define TILE_N 128
#define LDS_STRIDE 72           // 64 bf16 + 8 pad (144B, 16B aligned rows)

typedef __attribute__((ext_vector_type(16))) __bf16 bf16x16;
typedef __attribute__((ext_vector_type(2)))  __bf16 bf16x2;
typedef __attribute__((ext_vector_type(8)))  float  f32x8;
typedef __attribute__((ext_vector_type(4)))  float  f32x4;
typedef __attribute__((ext_vector_type(4)))  unsigned int u32x4;

union Frag {
    u32x4   q[2];
    bf16x16 v;
};

// Pack two f32 -> packed bf16 dword. Preferred: v_cvt_pk_bf16_f32 (if the
// toolchain exposes it). Otherwise: round-to-nearest bias + one v_perm_b32
// to gather the two high halves ({hi[31:16], lo[31:16]}) in a single VALU op.
__device__ __forceinline__ unsigned int pack2_bf16(float lo, float hi) {
#if __has_builtin(__builtin_amdgcn_cvt_pk_bf16_f32)
    bf16x2 p = __builtin_amdgcn_cvt_pk_bf16_f32(lo, hi);
    return __builtin_bit_cast(unsigned int, p);
#else
    unsigned int a = __builtin_bit_cast(unsigned int, lo) + 0x8000u;  // round-to-nearest
    unsigned int b = __builtin_bit_cast(unsigned int, hi) + 0x8000u;
#if __has_builtin(__builtin_amdgcn_perm)
    // result byte0=lo.b2, byte1=lo.b3, byte2=hi.b2, byte3=hi.b3
    return __builtin_amdgcn_perm(b, a, 0x07060302u);
#else
    return (a >> 16) | (b & 0xFFFF0000u);
#endif
#endif
}

__global__ __launch_bounds__(256)
void ternbin_wmma_kernel(const float* __restrict__ X,
                         const float* __restrict__ W,
                         const float* __restrict__ Bias,
                         float* __restrict__ OutX,
                         float* __restrict__ OutZ) {
    __shared__ alignas(16) unsigned short ldsA[2][TILE_M * LDS_STRIDE]; // x rows, bf16, row-major
    __shared__ alignas(16) unsigned short ldsB[2][TILE_N * LDS_STRIDE]; // W tile transposed: per-column contiguous K

    const int tid  = threadIdx.x;
    const int wave = tid >> 5;
    const int lane = tid & 31;
    const int wm   = wave & 3;    // wave position along M (32 rows each)
    const int wn   = wave >> 2;   // wave position along N (64 cols each)
    const int m0   = blockIdx.y * TILE_M;
    const int n0   = blockIdx.x * TILE_N;

    // --- global->LDS fill mapping ---
    // A: 128 rows x 64 K f32; thread covers 32 contiguous floats of one row-half
    const int arow  = tid >> 1;
    const int ahalf = (tid & 1) * 32;
    // B: 64 K rows x 128 N f32; thread owns a K-pair (2 rows) x 16 columns,
    //    so each column's two K values pack into one dword for LDS.
    const int bk = (tid >> 3) * 2;      // 0,2,...,62
    const int bn = (tid & 7) * 16;

    const float* gA  = X + (size_t)(m0 + arow) * DIN + ahalf;
    const float* gW0 = W + (size_t)bk * DOUT + (n0 + bn);
    const float* gW1 = gW0 + DOUT;

    f32x4 ra[8], rb0[4], rb1[4];

    auto load_tile = [&](int kt) {
        const f32x4* pa = (const f32x4*)(gA + (size_t)kt * KSTEP);
        #pragma unroll
        for (int i = 0; i < 8; ++i) ra[i] = pa[i];
        const f32x4* p0 = (const f32x4*)(gW0 + (size_t)kt * KSTEP * DOUT);
        const f32x4* p1 = (const f32x4*)(gW1 + (size_t)kt * KSTEP * DOUT);
        #pragma unroll
        for (int i = 0; i < 4; ++i) { rb0[i] = p0[i]; rb1[i] = p1[i]; }
    };

    auto store_tile = [&](int buf) {
        // A: 32 floats -> 16 packed-bf16 dword stores, contiguous
        float fa[32];
        #pragma unroll
        for (int i = 0; i < 8; ++i) {
            fa[4*i+0] = ra[i].x; fa[4*i+1] = ra[i].y;
            fa[4*i+2] = ra[i].z; fa[4*i+3] = ra[i].w;
        }
        unsigned int* da = (unsigned int*)&ldsA[buf][arow * LDS_STRIDE + ahalf];
        #pragma unroll
        for (int i = 0; i < 16; ++i)
            da[i] = pack2_bf16(fa[2*i], fa[2*i+1]);

        // B: transpose; pack (k, k+1) of each column into one dword store
        float f0[16], f1[16];
        #pragma unroll
        for (int i = 0; i < 4; ++i) {
            f0[4*i+0] = rb0[i].x; f0[4*i+1] = rb0[i].y;
            f0[4*i+2] = rb0[i].z; f0[4*i+3] = rb0[i].w;
            f1[4*i+0] = rb1[i].x; f1[4*i+1] = rb1[i].y;
            f1[4*i+2] = rb1[i].z; f1[4*i+3] = rb1[i].w;
        }
        #pragma unroll
        for (int j = 0; j < 16; ++j) {
            unsigned int* db = (unsigned int*)&ldsB[buf][(bn + j) * LDS_STRIDE + bk];
            *db = pack2_bf16(f0[j], f1[j]);
        }
    };

    // --- per-lane WMMA fragment addressing (ISA 16-bit layouts, wave32) ---
    const int fr = lane & 15;   // row (A) / col (B)
    const int fh = lane >> 4;   // lane-half selects K sub-range

    f32x8 acc[2][4] = {};

    auto compute = [&](int buf) {
        #pragma unroll
        for (int ko = 0; ko < KSTEP; ko += 32) {
            Frag a[2];
            #pragma unroll
            for (int t = 0; t < 2; ++t) {
                // A 16x32: lane<16 holds K{0..7,16..23}; lane>=16 holds K{8..15,24..31}
                const unsigned short* pa =
                    &ldsA[buf][(wm * 32 + t * 16 + fr) * LDS_STRIDE + ko + fh * 8];
                a[t].q[0] = *(const u32x4*)pa;
                a[t].q[1] = *(const u32x4*)(pa + 16);
            }
            #pragma unroll
            for (int n = 0; n < 4; ++n) {
                Frag b;
                // B 32x16: lane holds column fr; lane<16 holds K{0..7,16..23}; lane>=16 holds K{8..15,24..31}
                const unsigned short* pb =
                    &ldsB[buf][(wn * 64 + n * 16 + fr) * LDS_STRIDE + ko + fh * 8];
                b.q[0] = *(const u32x4*)pb;
                b.q[1] = *(const u32x4*)(pb + 16);
                #pragma unroll
                for (int t = 0; t < 2; ++t)
                    acc[t][n] = __builtin_amdgcn_wmma_f32_16x16x32_bf16(
                        false, a[t].v, false, b.v, (short)0, acc[t][n], false, false);
            }
        }
    };

    // --- main K loop: double-buffered LDS, global loads overlapped with WMMA ---
    load_tile(0);
    store_tile(0);
    __syncthreads();

    for (int kt = 0; kt < NK; ++kt) {
        const int cur = kt & 1;
        if (kt + 1 < NK) {
            load_tile(kt + 1);
            if (kt + 2 < NK) {
                __builtin_prefetch(gA  + (size_t)(kt + 2) * KSTEP, 0, 1);
                __builtin_prefetch(gW0 + (size_t)(kt + 2) * KSTEP * DOUT, 0, 1);
            }
        }
        compute(cur);
        __syncthreads();                  // all waves done reading buf cur^1 last iter
        if (kt + 1 < NK) {
            store_tile(cur ^ 1);
            __syncthreads();              // fill visible before next compute
        }
    }

    // --- epilogue: bias add, write z and sign(z) ---
    // C/D layout: VGPR r -> M = r (lanes 0-15) or r+8 (lanes 16-31), N = lane&15
    #pragma unroll
    for (int t = 0; t < 2; ++t) {
        #pragma unroll
        for (int n = 0; n < 4; ++n) {
            const int col = n0 + wn * 64 + n * 16 + fr;
            const float bias = Bias[col];
            #pragma unroll
            for (int r = 0; r < 8; ++r) {
                const int row = m0 + wm * 32 + t * 16 + fh * 8 + r;
                const size_t idx = (size_t)row * DOUT + col;
                const float z = acc[t][n][r] + bias;
                OutZ[idx] = z;
                OutX[idx] = (z >= 0.0f) ? 1.0f : -1.0f;
            }
        }
    }
}

extern "C" void kernel_launch(void* const* d_in, const int* in_sizes, int n_in,
                              void* d_out, int out_size, void* d_ws, size_t ws_size,
                              hipStream_t stream) {
    const float* X    = (const float*)d_in[0];   // [8192, 4096] f32
    const float* W    = (const float*)d_in[1];   // [4096, 4096] f32 (ternary values)
    const float* Bias = (const float*)d_in[2];   // [4096] f32
    float* outX = (float*)d_out;                             // sign output first
    float* outZ = outX + (size_t)B_ROWS * DOUT;              // then pre-activation z

    dim3 grid(DOUT / TILE_N, B_ROWS / TILE_M);   // 32 x 64 workgroups
    dim3 block(256);                             // 8 waves (wave32)
    ternbin_wmma_kernel<<<grid, block, 0, stream>>>(X, W, Bias, outX, outZ);
}
